// BertCacheAttention_78752520339945
// MI455X (gfx1250) — matrix operand order repaired and checked by reading, not compiled, not run on hardware
//
#include <hip/hip_runtime.h>

// ---------------------------------------------------------------------------
// Fixed problem shape (from reference setup_inputs)
// ---------------------------------------------------------------------------
#define BATCH 240
#define SEQ   64
#define HDIM  768
#define NH    12
#define HD    64
#define L0    384
#define L1    128
#define LTOT  (L0 + L1 + SEQ)   // 576
#define BS0   8                 // c0 cache batch = BATCH / slot_dim
#define MTOT  (BATCH * SEQ)     // 15360 rows for the QKV GEMM
#define OUT0  (BATCH * SEQ * HDIM)  // elements of first output tensor
#define KVPITCH 68              // LDS row pitch (floats) for staged K/V tiles

typedef __bf16 bf16_t;
typedef __attribute__((ext_vector_type(16))) __bf16 v16bf;
typedef __attribute__((ext_vector_type(8)))  float  v8f;
typedef __attribute__((ext_vector_type(4)))  int    v4i;

typedef __attribute__((address_space(1))) v4i* gptr_v4i;
typedef __attribute__((address_space(3))) v4i* lptr_v4i;

#if __has_builtin(__builtin_amdgcn_global_load_async_to_lds_b128) && \
    __has_builtin(__builtin_amdgcn_s_wait_asynccnt)
#define HAS_ASYNC_LDS 1
#else
#define HAS_ASYNC_LDS 0
#endif

static __device__ __forceinline__ v8f wmma_bf16(v16bf a, v16bf b, v8f c) {
  // D = A(16x32 bf16) x B(32x16 bf16) + C(16x16 f32)
  return __builtin_amdgcn_wmma_f32_16x16x32_bf16(
      /*neg_a=*/false, a, /*neg_b=*/false, b,
      /*c_mod=*/(short)0, c, /*reuse_a=*/false, /*reuse_b=*/false);
}

// 16B global->LDS copy: async DMA path on CDNA5, plain copy fallback
static __device__ __forceinline__ void async_copy16(const float* g, float* l) {
#if HAS_ASYNC_LDS
  __builtin_amdgcn_global_load_async_to_lds_b128(
      (gptr_v4i)g, (lptr_v4i)l, /*imm offset=*/0, /*cpol=*/0);
#else
  *(float4*)l = *(const float4*)g;
#endif
}

// fp32 -> bf16 B-fragment: elements 0..7 from p[0..7], 8..15 from p[16..23]
static __device__ __forceinline__ v16bf load_b16_contig(const float* __restrict__ p) {
  float4 a = *(const float4*)(p + 0);
  float4 b = *(const float4*)(p + 4);
  float4 c = *(const float4*)(p + 16);
  float4 d = *(const float4*)(p + 20);
  v16bf r;
  r[0]=(bf16_t)a.x; r[1]=(bf16_t)a.y; r[2]=(bf16_t)a.z; r[3]=(bf16_t)a.w;
  r[4]=(bf16_t)b.x; r[5]=(bf16_t)b.y; r[6]=(bf16_t)b.z; r[7]=(bf16_t)b.w;
  r[8]=(bf16_t)c.x; r[9]=(bf16_t)c.y; r[10]=(bf16_t)c.z; r[11]=(bf16_t)c.w;
  r[12]=(bf16_t)d.x; r[13]=(bf16_t)d.y; r[14]=(bf16_t)d.z; r[15]=(bf16_t)d.w;
  return r;
}

// A-fragment: elements 0..7 from p[0..7], elements 8..15 from p[16..23]
// (ISA 16-bit A 16x32 layout; caller adds the per-lane kbase)
static __device__ __forceinline__ v16bf load_a16(const float* __restrict__ p) {
  float4 a = *(const float4*)(p + 0);
  float4 b = *(const float4*)(p + 4);
  float4 c = *(const float4*)(p + 16);
  float4 d = *(const float4*)(p + 20);
  v16bf r;
  r[0]=(bf16_t)a.x; r[1]=(bf16_t)a.y; r[2]=(bf16_t)a.z; r[3]=(bf16_t)a.w;
  r[4]=(bf16_t)b.x; r[5]=(bf16_t)b.y; r[6]=(bf16_t)b.z; r[7]=(bf16_t)b.w;
  r[8]=(bf16_t)c.x; r[9]=(bf16_t)c.y; r[10]=(bf16_t)c.z; r[11]=(bf16_t)c.w;
  r[12]=(bf16_t)d.x; r[13]=(bf16_t)d.y; r[14]=(bf16_t)d.z; r[15]=(bf16_t)d.w;
  return r;
}

// B-fragment with strided K (rows of a row-major [K, pitch] LDS tile)
static __device__ __forceinline__ v16bf load_b16_strided(const float* __restrict__ p, int stride) {
  v16bf r;
#pragma unroll
  for (int e = 0; e < 16; ++e) r[e] = (bf16_t)p[(e + (e & 8)) * stride];
  return r;
}

// ---------------------------------------------------------------------------
// Kernel 1: out[m,n] = (X[m,:] . W[n,:]) + bias[n], scaled, written in head
// layout [B, NH, S, HD].  One wave per 16x16 output tile, 4 waves per block.
// ---------------------------------------------------------------------------
__global__ __launch_bounds__(128)
void qkv_gemm_kernel(const float* __restrict__ X,     // [MTOT, HDIM]
                     const float* __restrict__ W,     // [HDIM, HDIM] (row n, col k)
                     const float* __restrict__ bias,  // [HDIM]
                     float* __restrict__ dst,         // [BATCH, NH, SEQ, HD]
                     float scale)
{
  const int lane = threadIdx.x & 31;
  const int wave = threadIdx.x >> 5;
  const int tile = blockIdx.x * 4 + wave;         // 960 * 48 tiles
  const int nt = tile % (HDIM / 16);
  const int mt = tile / (HDIM / 16);
  const int hl = lane & 15;
  const int row  = mt * 16 + hl;                  // A row for this lane
  const int coln = nt * 16 + hl;                  // B col for this lane
  const int ka = (lane >> 4) * 8;                 // A-frag K base
  const int kb = (lane >> 4) * 8;                 // B-frag K base

  const float* xrow = X + (size_t)row  * HDIM;
  const float* wrow = W + (size_t)coln * HDIM;

  v8f acc = {};
#pragma unroll 4
  for (int k0 = 0; k0 < HDIM; k0 += 32) {
    v16bf a = load_a16(xrow + k0 + ka);
    v16bf b = load_b16_contig(wrow + k0 + kb);
    acc = wmma_bf16(a, b, acc);
  }

  const float bn = bias[coln];
  const int mrow_hi = (lane >> 4) << 3;           // +8 rows for upper half-wave
  const int h = coln >> 6;                        // head index
  const int d = coln & 63;
#pragma unroll
  for (int r = 0; r < 8; ++r) {
    const int m = mt * 16 + r + mrow_hi;          // global token row
    const int bb = m >> 6;                        // batch
    const int s  = m & 63;                        // seq pos
    dst[(((size_t)bb * NH + h) * SEQ + s) * HD + d] = (acc[r] + bn) * scale;
  }
}

// ---------------------------------------------------------------------------
// Kernel 2: per-(b,h) flash attention over 9 key tiles of 64
//   tiles 0-5: c0 cache (batch index b%8), 6-7: c1 cache, 8: self k/v
// 4 waves * 32 lanes; wave w owns query rows 16w..16w+15.
// K/V tiles are staged global->LDS with async-to-LDS DMA, double buffered:
// tile kt+1 streams in (ASYNCcnt) while WMMAs consume tile kt from LDS.
// ---------------------------------------------------------------------------
__global__ __launch_bounds__(128)
void attn_kernel(const float* __restrict__ q,     // [BATCH, NH, SEQ, HD], pre-scaled 1/8
                 const float* __restrict__ mask,  // [BATCH, LTOT]
                 const float* __restrict__ c0k,   // [BS0, NH, L0, HD]
                 const float* __restrict__ c0v,
                 const float* __restrict__ c1k,   // [BATCH, NH, L1, HD]
                 const float* __restrict__ c1v,
                 const float* __restrict__ kself, // [BATCH, NH, SEQ, HD]
                 const float* __restrict__ vself,
                 float* __restrict__ out)         // [BATCH, SEQ, HDIM]
{
  __shared__ float  kbuf[2][64][KVPITCH];         // staged K tiles (fp32), 2 buffers
  __shared__ float  vbuf[2][64][KVPITCH];         // staged V tiles
  __shared__ bf16_t p_lds[4][16][64];             // per-wave P strip (bf16)

  const int tid  = threadIdx.x;
  const int lane = tid & 31;
  const int wave = tid >> 5;
  const int bh = blockIdx.x;
  const int b = bh / NH, h = bh % NH;
  const int hl = lane & 15;
  const int ka = (lane >> 4) * 8;
  const int kb = (lane >> 4) * 8;
  const int rhi = (lane >> 4) << 3;               // row offset for upper half-wave

  // tile -> global source pointers
  auto tile_src = [&](int kt, const float*& Ks, const float*& Vs) {
    if (kt < 6) {
      const size_t base = (((size_t)(b % BS0) * NH + h) * L0 + kt * 64) * HD;
      Ks = c0k + base;  Vs = c0v + base;
    } else if (kt < 8) {
      const size_t base = (((size_t)b * NH + h) * L1 + (kt - 6) * 64) * HD;
      Ks = c1k + base;  Vs = c1v + base;
    } else {
      const size_t base = (((size_t)b * NH + h) * SEQ) * HD;
      Ks = kself + base;  Vs = vself + base;
    }
  };

  // stage one 64x64 fp32 K tile + V tile into LDS buffer bi (all 128 threads,
  // 16 async b128 issues per thread per tile-pair)
  auto stage_tile = [&](int kt, int bi) {
    const float *Ks, *Vs;
    tile_src(kt, Ks, Vs);
#pragma unroll
    for (int i = 0; i < 8; ++i) {
      const int flat = (i * 128 + tid) * 4;       // float index within 64x64 tile
      const int r = flat >> 6;
      const int c = flat & 63;
      async_copy16(Ks + flat, &kbuf[bi][r][c]);
      async_copy16(Vs + flat, &vbuf[bi][r][c]);
    }
  };

  // Q A-fragments for this wave's 16 rows (K = head dim, split 0-31 / 32-63)
  const float* qp = q + (((size_t)b * NH + h) * SEQ + wave * 16 + hl) * HD;
  const v16bf qa0 = load_a16(qp + ka);
  const v16bf qa1 = load_a16(qp + 32 + ka);

  float mrow[8], lrow[8];
  v8f o[4] = {v8f{}, v8f{}, v8f{}, v8f{}};
#pragma unroll
  for (int r = 0; r < 8; ++r) { mrow[r] = -__builtin_inff(); lrow[r] = 0.0f; }

  const float* mp = mask + (size_t)b * LTOT;

  stage_tile(0, 0);                               // prefetch first tile

  for (int kt = 0; kt < 9; ++kt) {
    const int bi = kt & 1;
    if (kt < 8) {
      stage_tile(kt + 1, bi ^ 1);                 // overlap next tile's DMA
#if HAS_ASYNC_LDS
      __builtin_amdgcn_s_wait_asynccnt(16);       // oldest 16 (tile kt) landed
#endif
    } else {
#if HAS_ASYNC_LDS
      __builtin_amdgcn_s_wait_asynccnt(0);
#endif
    }
    __syncthreads();                              // all waves' staging visible

    // ---- S tile strip: 16 rows x 64 cols, as 4 C fragments --------------
    v8f c[4];
#pragma unroll
    for (int j = 0; j < 4; ++j) {
      const float* kr = &kbuf[bi][j * 16 + hl][0];  // key row = this lane's col
      v16bf b0 = load_b16_contig(kr + kb);          // d = 0..31
      v16bf b1 = load_b16_contig(kr + 32 + kb);     // d = 32..63
      v8f z = {};
      z = wmma_bf16(qa0, b0, z);
      z = wmma_bf16(qa1, b1, z);
      const float mk = mp[kt * 64 + j * 16 + hl];   // additive mask (per col)
#pragma unroll
      for (int e = 0; e < 8; ++e) z[e] += mk;
      c[j] = z;
    }

    // ---- online softmax: row max across 64 cols -------------------------
    float tm[8];
#pragma unroll
    for (int r = 0; r < 8; ++r)
      tm[r] = fmaxf(fmaxf(c[0][r], c[1][r]), fmaxf(c[2][r], c[3][r]));
#pragma unroll
    for (int off = 1; off <= 8; off <<= 1)
#pragma unroll
      for (int r = 0; r < 8; ++r)
        tm[r] = fmaxf(tm[r], __shfl_xor(tm[r], off, 32));  // stays in 16-lane half

    float sc[8];
#pragma unroll
    for (int r = 0; r < 8; ++r) {
      const float mn = fmaxf(mrow[r], tm[r]);
      sc[r] = __expf(mrow[r] - mn);
      mrow[r] = mn;
    }

    float ts[8];
#pragma unroll
    for (int r = 0; r < 8; ++r) ts[r] = 0.0f;
#pragma unroll
    for (int j = 0; j < 4; ++j)
#pragma unroll
      for (int e = 0; e < 8; ++e) {
        const float p = __expf(c[j][e] - mrow[e]);
        c[j][e] = p;
        ts[e] += p;
      }
#pragma unroll
    for (int off = 1; off <= 8; off <<= 1)
#pragma unroll
      for (int r = 0; r < 8; ++r)
        ts[r] += __shfl_xor(ts[r], off, 32);
#pragma unroll
    for (int r = 0; r < 8; ++r) lrow[r] = lrow[r] * sc[r] + ts[r];

    // rescale running output accumulators
#pragma unroll
    for (int j = 0; j < 4; ++j)
#pragma unroll
      for (int e = 0; e < 8; ++e) o[j][e] *= sc[e];

    // ---- transpose P (C layout -> A layout) through LDS -----------------
#pragma unroll
    for (int j = 0; j < 4; ++j)
#pragma unroll
      for (int e = 0; e < 8; ++e)
        p_lds[wave][e + rhi][j * 16 + hl] = (bf16_t)c[j][e];
    __syncthreads();

    v16bf pa0, pa1;
    const bf16_t* pp = &p_lds[wave][hl][0];
#pragma unroll
    for (int e = 0; e < 8; ++e) {
      pa0[e]     = pp[ka + e];
      pa0[8 + e] = pp[16 + ka + e];
      pa1[e]     = pp[32 + ka + e];
      pa1[8 + e] = pp[48 + ka + e];
    }

    // ---- O += P @ V (V B-frags from staged LDS tile) ---------------------
#pragma unroll
    for (int j = 0; j < 4; ++j) {
      const float* vp = &vbuf[bi][0][j * 16 + hl];           // n = lane's d col
      v16bf v0 = load_b16_strided(vp + kb * KVPITCH, KVPITCH);        // kk 0..31
      v16bf v1 = load_b16_strided(vp + (32 + kb) * KVPITCH, KVPITCH); // kk 32..63
      o[j] = wmma_bf16(pa0, v0, o[j]);
      o[j] = wmma_bf16(pa1, v1, o[j]);
    }
    __syncthreads();   // protects p_lds AND kv buffers before next staging
  }

  // ---- normalize and store: out[b, s, h*64 + d] -------------------------
  float inv[8];
#pragma unroll
  for (int r = 0; r < 8; ++r) inv[r] = 1.0f / lrow[r];

  float* op = out + (size_t)b * SEQ * HDIM + h * HD;
#pragma unroll
  for (int j = 0; j < 4; ++j)
#pragma unroll
    for (int e = 0; e < 8; ++e) {
      const int s = wave * 16 + e + rhi;
      op[(size_t)s * HDIM + j * 16 + hl] = o[j][e] * inv[e];
    }
}

// ---------------------------------------------------------------------------
extern "C" void kernel_launch(void* const* d_in, const int* in_sizes, int n_in,
                              void* d_out, int out_size, void* d_ws, size_t ws_size,
                              hipStream_t stream) {
  (void)in_sizes; (void)n_in; (void)out_size; (void)ws_size;
  const float* hid  = (const float*)d_in[0];
  const float* mask = (const float*)d_in[1];
  const float* Wq   = (const float*)d_in[2];
  const float* bq   = (const float*)d_in[3];
  const float* Wk   = (const float*)d_in[4];
  const float* bk   = (const float*)d_in[5];
  const float* Wv   = (const float*)d_in[6];
  const float* bv   = (const float*)d_in[7];
  const float* c0k  = (const float*)d_in[8];
  const float* c0v  = (const float*)d_in[9];
  const float* c1k  = (const float*)d_in[10];
  const float* c1v  = (const float*)d_in[11];
  // d_in[12] = slot_dim (30) -> bs = 8, baked into BS0

  float* out  = (float*)d_out;
  float* kout = out + OUT0;              // k heads output
  float* vout = out + 2 * (size_t)OUT0;  // v heads output
  float* qws  = (float*)d_ws;            // q heads scratch, pre-scaled by 1/sqrt(HD)

  const int gemm_blocks = (MTOT / 16) * (HDIM / 16) / 4;  // 11520
  qkv_gemm_kernel<<<gemm_blocks, 128, 0, stream>>>(hid, Wq, bq, qws, 0.125f);
  qkv_gemm_kernel<<<gemm_blocks, 128, 0, stream>>>(hid, Wk, bk, kout, 1.0f);
  qkv_gemm_kernel<<<gemm_blocks, 128, 0, stream>>>(hid, Wv, bv, vout, 1.0f);

  attn_kernel<<<BATCH * NH, 128, 0, stream>>>(qws, mask, c0k, c0v, c1k, c1v,
                                              kout, vout, out);
}
